// GlobalChannelAttention_25692494365580
// MI455X (gfx1250) — matrix operand order repaired and checked by reading, not compiled, not run on hardware
//
#include <hip/hip_runtime.h>
#include <math.h>

typedef __attribute__((ext_vector_type(16))) __bf16        v16bf;
typedef __attribute__((ext_vector_type(2)))  __bf16        v2bf;
typedef __attribute__((ext_vector_type(8)))  float         v8f;
typedef __attribute__((ext_vector_type(4)))  unsigned int  u32x4;
typedef __attribute__((ext_vector_type(2)))  unsigned int  u32x2;
typedef __attribute__((ext_vector_type(4)))  float         f32x4;

// TDM descriptor group types (probe-verified builtin signatures)
typedef __attribute__((ext_vector_type(4))) unsigned int tdm_g0_t;
typedef __attribute__((ext_vector_type(8))) int          tdm_g1_t;
typedef __attribute__((ext_vector_type(4))) int          tdm_g2_t;

struct Pair { u32x4 lo, hi; };   // 32B: one full 16-bf16 fragment

static constexpr int Cc = 256;     // channels
static constexpr int Nn = 4096;    // H*W
static constexpr int Bb = 16;      // batch

// native converts (v_cvt_pk_bf16_f32 path)
__device__ inline unsigned short f2bf(float f) {
  __bf16 h = (__bf16)f;
  return __builtin_bit_cast(unsigned short, h);
}

__device__ inline unsigned int pack2(float a, float b) {
  v2bf t; t[0] = (__bf16)a; t[1] = (__bf16)b;
  return __builtin_bit_cast(unsigned int, t);
}

__device__ inline v16bf frag2(u32x4 lo, u32x4 hi) {
  Pair p{lo, hi};
  return __builtin_bit_cast(v16bf, p);
}

// 2D bf16 tile DMA: global (row-major, stride in elements) -> LDS (row-major)
__device__ inline void tdm_load_2d_bf16(unsigned lds_off, const void* gaddr,
                                        unsigned tile_rows, unsigned tile_cols,
                                        unsigned tensor_rows, unsigned tensor_cols,
                                        unsigned row_stride_elems) {
  unsigned long long ga = (unsigned long long)gaddr;
  tdm_g0_t g0;
  g0[0] = 1u;                                              // count=1 (valid, user)
  g0[1] = lds_off;                                         // lds_addr (bytes)
  g0[2] = (unsigned)(ga & 0xFFFFFFFFu);                    // global_addr[31:0]
  g0[3] = (unsigned)((ga >> 32) & 0x1FFFFFFu) | (2u << 30);// global_addr[56:32] | type=2
  tdm_g1_t g1;
  g1[0] = (int)(1u << 16);                                 // data_size=1 (2 bytes)
  g1[1] = (int)((tensor_cols & 0xFFFFu) << 16);            // tensor_dim0[15:0]
  g1[2] = (int)((tensor_cols >> 16) | ((tensor_rows & 0xFFFFu) << 16));
  g1[3] = (int)((tensor_rows >> 16) | ((tile_cols & 0xFFFFu) << 16)); // tile_dim0
  g1[4] = (int)(tile_rows & 0xFFFFu);                      // tile_dim1 (tile_dim2=0)
  g1[5] = (int)row_stride_elems;                           // tensor_dim0_stride[31:0]
  g1[6] = 0;
  g1[7] = 0;
  tdm_g2_t g2 = {};
  tdm_g2_t g3 = {};
#if defined(__clang_major__) && (__clang_major__ >= 23)
  tdm_g1_t g4 = {};
  __builtin_amdgcn_tensor_load_to_lds(g0, g1, g2, g3, g4, 0);
#else
  __builtin_amdgcn_tensor_load_to_lds(g0, g1, g2, g3, 0);
#endif
}

// ---------------------------------------------------------------------------
// Kernel 1: fused 1x1-conv projections. out_p = W_p * X + b_p, p in {v,q,k}
// X: f32 [B, C, N]; W: f32 [C,C]; outputs bf16 [B, C, N]
// grid = (N/64, B, 3), block = 256 (8 waves); wave -> 32 M rows, block -> 64 N cols
// ---------------------------------------------------------------------------
__global__ void __launch_bounds__(256) proj_kernel(
    const float* __restrict__ x,
    const float* __restrict__ w1, const float* __restrict__ b1,
    const float* __restrict__ w2, const float* __restrict__ b2,
    const float* __restrict__ w3, const float* __restrict__ b3,
    unsigned short* __restrict__ vbf,
    unsigned short* __restrict__ qbf,
    unsigned short* __restrict__ kbf)
{
  const int p = blockIdx.z;
  const float* W      = (p == 0) ? w1 : (p == 1) ? w2 : w3;
  const float* bias   = (p == 0) ? b1 : (p == 1) ? b2 : b3;
  unsigned short* dst = (p == 0) ? vbf : (p == 1) ? qbf : kbf;

  const int b    = blockIdx.y;
  const int n0   = blockIdx.x * 64;
  const int lane = threadIdx.x & 31;
  const int wv   = threadIdx.x >> 5;
  const int m0   = wv * 32;
  const int col  = lane & 15;
  const int hi   = lane >> 4;          // which K-half of the wave
  const int kbA  = hi * 8;             // A frag K base (ISA 16-bit A layout)
  const int kbB  = hi * 8;             // B frag K base (same table order as A)

  __shared__ unsigned short xs[64][40];   // [local col][k], 80B rows (16B aligned)

  v8f zero = {};
  v8f acc[2][4];
#pragma unroll
  for (int t = 0; t < 2; ++t)
#pragma unroll
    for (int j = 0; j < 4; ++j) acc[t][j] = zero;

  const float* xb = x + (size_t)b * Cc * Nn;

  // staging assignment: 4 cols x 2 rows per thread -> packed ds_store_b32
  const int scst = (threadIdx.x & 15) * 4;   // col start
  const int srow = (threadIdx.x >> 4) * 2;   // row pair

  for (int k0 = 0; k0 < Cc; k0 += 32) {
    // stage X[k0:k0+32, n0:n0+64] transposed + converted to bf16 in LDS
    {
      const float* s0 = xb + (size_t)(k0 + srow) * Nn + n0 + scst;
      f32x4 a = *reinterpret_cast<const f32x4*>(s0);
      f32x4 c = *reinterpret_cast<const f32x4*>(s0 + Nn);
#pragma unroll
      for (int i = 0; i < 4; ++i)   // one v_cvt_pk_bf16_f32 + one ds_store_b32 each
        *reinterpret_cast<unsigned int*>(&xs[scst + i][srow]) = pack2(a[i], c[i]);
    }
    __syncthreads();

    // A fragments: W rows, f32 -> packed bf16 dwords
    v16bf afr[2];
#pragma unroll
    for (int t = 0; t < 2; ++t) {
      const float* wr = W + (size_t)(m0 + t * 16 + col) * Cc + k0 + kbA;
      f32x4 a0 = *reinterpret_cast<const f32x4*>(wr);
      f32x4 a1 = *reinterpret_cast<const f32x4*>(wr + 4);
      f32x4 a2 = *reinterpret_cast<const f32x4*>(wr + 16);
      f32x4 a3 = *reinterpret_cast<const f32x4*>(wr + 20);
      u32x4 lo, hv;
      lo[0] = pack2(a0[0], a0[1]); lo[1] = pack2(a0[2], a0[3]);
      lo[2] = pack2(a1[0], a1[1]); lo[3] = pack2(a1[2], a1[3]);
      hv[0] = pack2(a2[0], a2[1]); hv[1] = pack2(a2[2], a2[3]);
      hv[2] = pack2(a3[0], a3[1]); hv[3] = pack2(a3[2], a3[3]);
      afr[t] = frag2(lo, hv);
    }

    // B fragments straight from LDS (two ds_load_b128 each) + WMMA
#pragma unroll
    for (int j = 0; j < 4; ++j) {
      const unsigned short* bp = &xs[j * 16 + col][kbB];
      v16bf bfr = frag2(*reinterpret_cast<const u32x4*>(bp),
                        *reinterpret_cast<const u32x4*>(bp + 16));
#pragma unroll
      for (int t = 0; t < 2; ++t)
        acc[t][j] = __builtin_amdgcn_wmma_f32_16x16x32_bf16(
            false, afr[t], false, bfr, (short)0, acc[t][j], false, false);
    }
    __syncthreads();
  }

  // bias add + bf16 store
#pragma unroll
  for (int t = 0; t < 2; ++t) {
#pragma unroll
    for (int r = 0; r < 8; ++r) {
      int m = m0 + t * 16 + r + hi * 8;
      float bv = bias[m];
#pragma unroll
      for (int j = 0; j < 4; ++j)
        dst[((size_t)b * Cc + m) * Nn + n0 + j * 16 + col] = f2bf(acc[t][j][r] + bv);
    }
  }
}

// ---------------------------------------------------------------------------
// Kernel 2: logits[b][c][d] = sum_n q[b][c][n] * k[b][d][n]
// Double-buffered TDM pipeline: wave0 waits for panel i, issues panel i+1 into
// the other LDS buffer, then all 8 waves compute panel i while the TDM engine
// streams panel i+1. grid = (16, 2, B), block = 256.
// ---------------------------------------------------------------------------
__global__ void __launch_bounds__(256) qk_kernel(
    const unsigned short* __restrict__ q,
    const unsigned short* __restrict__ k,
    float* __restrict__ logits)
{
  constexpr int KB = 64;                        // K panel (elements)
  constexpr int NP = Nn / KB;                   // panels
  const int b    = blockIdx.z;
  const int c0   = blockIdx.x * 16;
  const int d0b  = blockIdx.y * 128;            // 8 d-tiles per block
  const int wv   = threadIdx.x >> 5;
  const int lane = threadIdx.x & 31;
  const int col  = lane & 15;
  const int hi   = lane >> 4;
  const int kbA  = hi * 8;
  const int kbB  = hi * 8;

  __shared__ unsigned short qs[2][16 * KB];     // 2 x 2 KB
  __shared__ unsigned short ks[2][128 * KB];    // 2 x 16 KB

  const unsigned short* qbase = q + ((size_t)b * Cc + c0) * Nn;
  const unsigned short* kbase = k + ((size_t)b * Cc + d0b) * Nn;

  if (threadIdx.x == 0) {
    tdm_load_2d_bf16((unsigned)(size_t)&qs[0][0], qbase, 16, KB, Cc, Nn, Nn);
    tdm_load_2d_bf16((unsigned)(size_t)&ks[0][0], kbase, 128, KB, Cc, Nn, Nn);
  }

  v8f acc = {};
  for (int i = 0; i < NP; ++i) {
    const int cur = i & 1;
    if (threadIdx.x == 0) {
      __builtin_amdgcn_s_wait_tensorcnt(0);     // panel i landed
      if (i + 1 < NP) {                         // stream panel i+1 during compute
        const int nxt = cur ^ 1;
        tdm_load_2d_bf16((unsigned)(size_t)&qs[nxt][0], qbase + (i + 1) * KB,
                         16, KB, Cc, Nn, Nn);
        tdm_load_2d_bf16((unsigned)(size_t)&ks[nxt][0], kbase + (i + 1) * KB,
                         128, KB, Cc, Nn, Nn);
      }
    }
    __syncthreads();                            // panel i visible to all waves

#pragma unroll
    for (int kk = 0; kk < KB; kk += 32) {
      const unsigned short* ap = &qs[cur][col * KB + kk + kbA];
      v16bf afr = frag2(*reinterpret_cast<const u32x4*>(ap),
                        *reinterpret_cast<const u32x4*>(ap + 16));
      const unsigned short* bp = &ks[cur][(wv * 16 + col) * KB + kk + kbB];
      v16bf bfr = frag2(*reinterpret_cast<const u32x4*>(bp),
                        *reinterpret_cast<const u32x4*>(bp + 16));
      acc = __builtin_amdgcn_wmma_f32_16x16x32_bf16(
          false, afr, false, bfr, (short)0, acc, false, false);
    }
    __syncthreads();                            // done reading buf before reuse
  }

  float* Lg = logits + (size_t)b * Cc * Cc;
  const int d0 = d0b + wv * 16;
#pragma unroll
  for (int r = 0; r < 8; ++r)
    Lg[(size_t)(c0 + r + hi * 8) * Cc + d0 + col] = acc[r];
}

// ---------------------------------------------------------------------------
// Kernel 3: global softmax over C*C per batch; writes TRANSPOSED weights (bf16)
// so the final GEMM's A operand is row-contiguous. grid = B, block = 1024.
// ---------------------------------------------------------------------------
__global__ void __launch_bounds__(1024) softmax_kernel(
    const float* __restrict__ logits,
    unsigned short* __restrict__ wT)
{
  const int b = blockIdx.x;
  const float* L = logits + (size_t)b * Cc * Cc;
  __shared__ float red[1024];

  float m = -INFINITY;
  for (int i = threadIdx.x; i < Cc * Cc; i += 1024) m = fmaxf(m, L[i]);
  red[threadIdx.x] = m; __syncthreads();
  for (int s = 512; s > 0; s >>= 1) {
    if ((int)threadIdx.x < s) red[threadIdx.x] = fmaxf(red[threadIdx.x], red[threadIdx.x + s]);
    __syncthreads();
  }
  m = red[0]; __syncthreads();

  float sum = 0.f;
  for (int i = threadIdx.x; i < Cc * Cc; i += 1024) sum += __expf(L[i] - m);
  red[threadIdx.x] = sum; __syncthreads();
  for (int s = 512; s > 0; s >>= 1) {
    if ((int)threadIdx.x < s) red[threadIdx.x] += red[threadIdx.x + s];
    __syncthreads();
  }
  float inv = 1.0f / red[0];

  unsigned short* Wt = wT + (size_t)b * Cc * Cc;
  for (int i = threadIdx.x; i < Cc * Cc; i += 1024) {
    int c = i >> 8, d = i & 255;
    Wt[(size_t)d * Cc + c] = f2bf(__expf(L[i] - m) * inv);  // transposed
  }
}

// ---------------------------------------------------------------------------
// Kernel 4: out[b][d][n] = sum_c weight[c][d] * v[b][c][n]  (A = wT row-major)
// grid = (N/64, B), block = 256 (8 waves x 32 rows)
// ---------------------------------------------------------------------------
__global__ void __launch_bounds__(256) out_kernel(
    const unsigned short* __restrict__ wT,
    const unsigned short* __restrict__ v,
    float* __restrict__ out)
{
  const int b    = blockIdx.y;
  const int n0   = blockIdx.x * 64;
  const int wv   = threadIdx.x >> 5;
  const int lane = threadIdx.x & 31;
  const int m0   = wv * 32;
  const int col  = lane & 15;
  const int hi   = lane >> 4;

  __shared__ unsigned short vs[64][40];

  v8f zero = {};
  v8f acc[2][4];
#pragma unroll
  for (int t = 0; t < 2; ++t)
#pragma unroll
    for (int j = 0; j < 4; ++j) acc[t][j] = zero;

  const unsigned short* vb = v  + (size_t)b * Cc * Nn;
  const unsigned short* wb = wT + (size_t)b * Cc * Cc;

  const int scst = (threadIdx.x & 15) * 4;   // col start (4 cols x 2 rows / thread)
  const int srow = (threadIdx.x >> 4) * 2;   // row pair

  for (int k0 = 0; k0 < Cc; k0 += 32) {
    // stage v[k0:k0+32, n0:n0+64] transposed in LDS; packed b32 stores
    {
      const unsigned short* s0 = vb + (size_t)(k0 + srow) * Nn + n0 + scst;
      if (k0 + 32 < Cc) __builtin_prefetch(s0 + 32 * Nn, 0, 0);
      u32x2 a = *reinterpret_cast<const u32x2*>(s0);        // row r,  4 bf16
      u32x2 c = *reinterpret_cast<const u32x2*>(s0 + Nn);   // row r+1, 4 bf16
      unsigned m0v = (a[0] & 0xFFFFu)  | (c[0] << 16);
      unsigned m1v = (a[0] >> 16)      | (c[0] & 0xFFFF0000u);
      unsigned m2v = (a[1] & 0xFFFFu)  | (c[1] << 16);
      unsigned m3v = (a[1] >> 16)      | (c[1] & 0xFFFF0000u);
      *reinterpret_cast<unsigned int*>(&vs[scst + 0][srow]) = m0v;
      *reinterpret_cast<unsigned int*>(&vs[scst + 1][srow]) = m1v;
      *reinterpret_cast<unsigned int*>(&vs[scst + 2][srow]) = m2v;
      *reinterpret_cast<unsigned int*>(&vs[scst + 3][srow]) = m3v;
    }
    __syncthreads();

    v16bf afr[2];
#pragma unroll
    for (int t = 0; t < 2; ++t) {
      const unsigned short* wr = wb + (size_t)(m0 + t * 16 + col) * Cc + k0 + hi * 8;
      afr[t] = frag2(*reinterpret_cast<const u32x4*>(wr),
                     *reinterpret_cast<const u32x4*>(wr + 16));
    }

#pragma unroll
    for (int j = 0; j < 4; ++j) {
      const unsigned short* bp = &vs[j * 16 + col][hi * 8];
      v16bf bfr = frag2(*reinterpret_cast<const u32x4*>(bp),
                        *reinterpret_cast<const u32x4*>(bp + 16));
#pragma unroll
      for (int t = 0; t < 2; ++t)
        acc[t][j] = __builtin_amdgcn_wmma_f32_16x16x32_bf16(
            false, afr[t], false, bfr, (short)0, acc[t][j], false, false);
    }
    __syncthreads();
  }

  float* ob = out + (size_t)b * Cc * Nn;
#pragma unroll
  for (int t = 0; t < 2; ++t)
#pragma unroll
    for (int r = 0; r < 8; ++r)
#pragma unroll
      for (int j = 0; j < 4; ++j)
        ob[(size_t)(m0 + t * 16 + r + hi * 8) * Nn + n0 + j * 16 + col] = acc[t][j][r];
}

// ---------------------------------------------------------------------------
extern "C" void kernel_launch(void* const* d_in, const int* in_sizes, int n_in,
                              void* d_out, int out_size, void* d_ws, size_t ws_size,
                              hipStream_t stream) {
  (void)in_sizes; (void)n_in; (void)out_size; (void)ws_size;
  const float* x  = (const float*)d_in[0];
  const float* w1 = (const float*)d_in[1];
  const float* b1 = (const float*)d_in[2];
  const float* w2 = (const float*)d_in[3];
  const float* b2 = (const float*)d_in[4];
  const float* w3 = (const float*)d_in[5];
  const float* b3 = (const float*)d_in[6];

  // workspace: v_bf16 (32MB) | logits f32 (4MB) | weightT bf16 (2MB)
  unsigned short* vbf   = (unsigned short*)d_ws;
  float*          logit = (float*)((char*)d_ws + (size_t)32 * 1024 * 1024);
  unsigned short* wT    = (unsigned short*)((char*)d_ws + (size_t)36 * 1024 * 1024);

  // q/k bf16 temporarily live in d_out (exactly 64MB); overwritten by out_kernel
  unsigned short* qbf = (unsigned short*)d_out;
  unsigned short* kbf = qbf + (size_t)Bb * Cc * Nn;

  proj_kernel<<<dim3(Nn / 64, Bb, 3), 256, 0, stream>>>(
      x, w1, b1, w2, b2, w3, b3, vbf, qbf, kbf);
  qk_kernel<<<dim3(Cc / 16, 2, Bb), 256, 0, stream>>>(qbf, kbf, logit);
  softmax_kernel<<<Bb, 1024, 0, stream>>>(logit, wT);
  out_kernel<<<dim3(Nn / 64, Bb), 256, 0, stream>>>(wT, vbf, (float*)d_out);
}
